// TextLSTM_19069654794359
// MI455X (gfx1250) — matrix operand order repaired and checked by reading, not compiled, not run on hardware
//
#include <hip/hip_runtime.h>
#include <hip/hip_bf16.h>

// ---------------- model dims ----------------
#define HID    512
#define GATES  2048            // 4*HID
#define BSZ    64
#define TLEN   256
#define NCLS   50257
#define NCLSP  50304           // padded to multiple of 128
#define NWG    16              // persistent workgroups in the scan kernel

// LDS layout for the scan kernel
#define HROW_B   1040          // 512 bf16 + 16B pad  -> conflict-free ds_load_b128
#define HLDS_B   (64 * HROW_B) // 66560
#define GOFF_B   HLDS_B        // gates: 64 x 128 f32 = 32768
#define COFF_B   (GOFF_B + 64 * 128 * 4)
#define SMEM_B   (COFF_B + 64 * 32 * 4)   // + c-state 8192 => 107520 bytes

// ---------------- vector types ----------------
typedef __bf16 v16bf __attribute__((ext_vector_type(16)));
typedef __bf16 v8bf  __attribute__((ext_vector_type(8)));
typedef float  v8f   __attribute__((ext_vector_type(8)));
typedef float  v4f   __attribute__((ext_vector_type(4)));
typedef unsigned int u32x4 __attribute__((ext_vector_type(4)));
typedef int    i32x4 __attribute__((ext_vector_type(4)));
typedef int    i32x8 __attribute__((ext_vector_type(8)));

__device__ __forceinline__ v8f fzero8() {
  v8f z;
#pragma unroll
  for (int i = 0; i < 8; ++i) z[i] = 0.0f;
  return z;
}

__device__ __forceinline__ v16bf cat8(v8bf lo, v8bf hi) {
  return __builtin_shufflevector(lo, hi, 0,1,2,3,4,5,6,7,8,9,10,11,12,13,14,15);
}

__device__ __forceinline__ v8f wmma_bf16(v16bf a, v16bf b, v8f c) {
  return __builtin_amdgcn_wmma_f32_16x16x32_bf16(false, a, false, b, (short)0, c,
                                                 false, false);
}

__device__ __forceinline__ float sigmoidf_(float x) {
  return 1.0f / (1.0f + __expf(-x));
}

// A-matrix fragment (16x32 bf16) from a row-major bf16 matrix with row stride
// `ld` elements; lanes 0-15 carry K {0..15}, lanes
// 16-31 carry K {16..31}.
__device__ __forceinline__ v16bf load_afrag_g(const __bf16* __restrict__ base,
                                              int row0, int ld, int kc,
                                              int ln, int lh) {
  const __bf16* p = base + (size_t)(row0 + ln) * ld + kc + (lh << 4);
  v8bf lo = *(const v8bf*)p;
  v8bf hi = *(const v8bf*)(p + 8);
  return cat8(lo, hi);
}

// ---------------- init: barrier counters + h_0 = 0 blocks ----------------
__global__ __launch_bounds__(256)
void k_init(__bf16* __restrict__ h0seq, __bf16* __restrict__ h1seq,
            int* __restrict__ bar) {
  int tid = threadIdx.x;
  if (tid < 8) bar[tid] = 0;
  v8bf z;
#pragma unroll
  for (int i = 0; i < 8; ++i) z[i] = (__bf16)0.0f;
  for (int i = tid * 8; i < BSZ * HID; i += 256 * 8) {
    *(v8bf*)(h0seq + i) = z;
    *(v8bf*)(h1seq + i) = z;
  }
}

// ---------------- f32 -> bf16 cast with zero padding ----------------
__global__ __launch_bounds__(256)
void k_cast(const float* __restrict__ src, __bf16* __restrict__ dst,
            int n, int npad) {
  int gid = (blockIdx.x * 256 + threadIdx.x) * 8;
  if (gid >= npad) return;
  v8bf o;
  if (gid + 8 <= n) {
    v4f f0 = *(const v4f*)(src + gid);
    v4f f1 = *(const v4f*)(src + gid + 4);
#pragma unroll
    for (int i = 0; i < 4; ++i) { o[i] = (__bf16)f0[i]; o[4+i] = (__bf16)f1[i]; }
  } else {
#pragma unroll
    for (int i = 0; i < 8; ++i) {
      int e = gid + i;
      o[i] = (e < n) ? (__bf16)src[e] : (__bf16)0.0f;
    }
  }
  *(v8bf*)(dst + gid) = o;
}

// ---------------- bias sum: b_ih + b_hh ----------------
__global__ __launch_bounds__(256)
void k_bias(const float* __restrict__ a, const float* __restrict__ b,
            float* __restrict__ o) {
  int i0 = threadIdx.x * 8;
#pragma unroll
  for (int q = 0; q < 8; ++q) o[i0 + q] = a[i0 + q] + b[i0 + q];
}

// ---------------- embedding gather -> bf16, (t,b,E) row order ----------------
__global__ __launch_bounds__(256)
void k_embed(const int* __restrict__ X, const float* __restrict__ emb,
             __bf16* __restrict__ xo) {
  int gid = blockIdx.x * 256 + threadIdx.x;   // 16384*512/8 threads
  int r  = gid >> 6;                          // row = t*64 + b
  int c0 = (gid & 63) << 3;
  int t = r >> 6, b = r & 63;
  int tok = X[b * TLEN + t];
  const v4f* s = (const v4f*)(emb + (size_t)tok * HID + c0);
  v4f f0 = s[0], f1 = s[1];
  v8bf o;
#pragma unroll
  for (int i = 0; i < 4; ++i) { o[i] = (__bf16)f0[i]; o[4+i] = (__bf16)f1[i]; }
  *(v8bf*)(xo + (size_t)r * HID + c0) = o;
}

// ---------------- big input GEMM: xg = A(16384x512) * W^T + bias -> bf16 ----
// A row-major bf16, W row-major bf16 (2048x512), out (16384 x 2048) bf16.
__global__ __launch_bounds__(256)
void k_gemm16k(const __bf16* __restrict__ A, const __bf16* __restrict__ Wt,
               const float* __restrict__ bias, __bf16* __restrict__ out) {
  const int lane = threadIdx.x & 31, wv = threadIdx.x >> 5;
  const int ln = lane & 15, lh = lane >> 4;
  const int r0 = blockIdx.y * 64;
  const int n0 = blockIdx.x * 256 + wv * 32;

  v8f acc[4][2];
#pragma unroll
  for (int mt = 0; mt < 4; ++mt)
#pragma unroll
    for (int nt = 0; nt < 2; ++nt) acc[mt][nt] = fzero8();

  for (int kc = 0; kc < HID; kc += 32) {
    v16bf a[4];
#pragma unroll
    for (int mt = 0; mt < 4; ++mt)
      a[mt] = load_afrag_g(A, r0 + mt * 16, HID, kc, ln, lh);
    v16bf bb[2];
#pragma unroll
    for (int nt = 0; nt < 2; ++nt) {
      const __bf16* p = Wt + (size_t)(n0 + nt * 16 + ln) * HID + kc + (lh << 4);
      bb[nt] = *(const v16bf*)p;
    }
#pragma unroll
    for (int mt = 0; mt < 4; ++mt)
#pragma unroll
      for (int nt = 0; nt < 2; ++nt)
        acc[mt][nt] = wmma_bf16(a[mt], bb[nt], acc[mt][nt]);
  }
#pragma unroll
  for (int nt = 0; nt < 2; ++nt) {
    int n = n0 + nt * 16 + ln;
    float bs = bias[n];
#pragma unroll
    for (int mt = 0; mt < 4; ++mt) {
#pragma unroll
      for (int r = 0; r < 8; ++r) {
        int m = r0 + mt * 16 + lh * 8 + r;
        out[(size_t)m * GATES + n] = (__bf16)(acc[mt][nt][r] + bs);
      }
    }
  }
}

// ---------------- persistent LSTM scan (one layer) ----------------
// xg  : (T*64, 2048) bf16, row t*64+b        (precomputed x@w_ih^T + biases)
// hseq: (T+1, 64, 512) bf16; block 0 is zero; we write block t+1 each step
// whh : (2048, 512) bf16 row-major
__global__ __launch_bounds__(256, 1)
void k_scan(const __bf16* __restrict__ xg, __bf16* __restrict__ hseq,
            const __bf16* __restrict__ whh, int* __restrict__ bar) {
  extern __shared__ char smem[];
  char*  hlds = smem;
  float* gl   = (float*)(smem + GOFF_B);
  float* cst  = (float*)(smem + COFF_B);

  const int tid  = threadIdx.x;
  const int lane = tid & 31;
  const int wv   = tid >> 5;
  const int wg   = blockIdx.x;
  const int ln   = lane & 15;
  const int lh   = lane >> 4;

  for (int i = tid; i < BSZ * 32; i += 256) cst[i] = 0.0f;
  __syncthreads();

  const int m0    = (wv & 3) * 16;   // M tile rows
  const int jbase = (wv >> 2) * 4;   // first of 4 N-tiles (of 8 per block)

  // B-fragment base pointers: gate column for N-tile jt, lane column ln
  const __bf16* bptr[4];
#pragma unroll
  for (int q = 0; q < 4; ++q) {
    int jt = jbase + q;
    int gcol = (jt >> 1) * HID + wg * 32 + (jt & 1) * 16 + ln;
    bptr[q] = whh + (size_t)gcol * HID + (lh << 4);
  }

  const unsigned int hlds_base = (unsigned int)(uintptr_t)hlds; // LDS byte offset

  for (int t = 0; t < TLEN; ++t) {
    // ---- TDM: pull h_t (64x512 bf16, contiguous 64KB) into padded LDS rows
    if (wv == 0) {
      unsigned long long gaddr =
          (unsigned long long)(uintptr_t)(hseq + (size_t)t * BSZ * HID);
      unsigned long long lo = 1ull /*count=1*/ | ((unsigned long long)hlds_base << 32);
      unsigned long long hi = (gaddr & 0x01FFFFFFFFFFFFFFull) | (2ull << 62); // type=2
      u32x4 g0 = { (unsigned int)lo, (unsigned int)(lo >> 32),
                   (unsigned int)hi, (unsigned int)(hi >> 32) };
      i32x8 g1;
      // data_size=2B, pad_enable, pad_interval=256 DW (one 1024B row),
      // pad_amount=4 DW (16B)  -> LDS row stride 1040
      g1[0] = (1 << 16) | (1 << 20) | (7 << 22) | (3 << 25);
      g1[1] = (int)((HID & 0xFFFF) << 16);          // tensor_dim0 (bits 79:48)
      g1[2] = (int)((HID >> 16) | (BSZ << 16));     // | tensor_dim1 (111:80)
      g1[3] = (int)((BSZ >> 16) | (HID << 16));     // | tile_dim0  (127:112)
      g1[4] = BSZ;                                   // tile_dim1 (tile_dim2=0)
      g1[5] = HID;                                   // tensor_dim0_stride
      g1[6] = 0; g1[7] = 0;
      i32x4 gz4 = {0, 0, 0, 0};
      i32x8 gz8;
#pragma unroll
      for (int i = 0; i < 8; ++i) gz8[i] = 0;
      // 6-arg toolchain variant: (g0, g1, g2, g3, g4, cpol)
      __builtin_amdgcn_tensor_load_to_lds(g0, g1, gz4, gz4, gz8, 0);
      __builtin_amdgcn_s_wait_tensorcnt(0);
    }
    __syncthreads();

    // ---- GEMM: gates[64 x 128] = h_t @ w_hh^T (this wg's 4x32 gate slice)
    v8f acc[4];
#pragma unroll
    for (int q = 0; q < 4; ++q) acc[q] = fzero8();

    for (int kc = 0; kc < HID; kc += 32) {
      const char* ap = hlds + (m0 + ln) * HROW_B + ((kc + (lh << 4)) << 1);
      v8bf alo = *(const v8bf*)ap;
      v8bf ahi = *(const v8bf*)(ap + 16);
      v16bf afr = cat8(alo, ahi);
#pragma unroll
      for (int q = 0; q < 4; ++q) {
        v16bf bfr = *(const v16bf*)(bptr[q] + kc);
        acc[q] = wmma_bf16(afr, bfr, acc[q]);
      }
    }
#pragma unroll
    for (int q = 0; q < 4; ++q) {
      int jcol = (jbase + q) * 16 + ln;
#pragma unroll
      for (int r = 0; r < 8; ++r)
        gl[(m0 + lh * 8 + r) * 128 + jcol] = acc[q][r];
    }
    __syncthreads();

    // ---- elementwise LSTM cell for this wg's 32 h-columns
    {
      int e0 = tid * 8;
      int b  = e0 >> 5;
      int j0 = e0 & 31;
      const __bf16* xgrow = xg + ((size_t)(t * BSZ + b)) * GATES + wg * 32;
      const float*  grow  = gl + b * 128;
      float*        crow  = cst + b * 32;
      __bf16*       hrow  = hseq + ((size_t)((t + 1) * BSZ + b)) * HID + wg * 32;
#pragma unroll
      for (int q = 0; q < 8; ++q) {
        int j = j0 + q;
        float gi = grow[j]        + (float)xgrow[j];
        float gf = grow[32 + j]   + (float)xgrow[HID + j];
        float gg = grow[64 + j]   + (float)xgrow[2 * HID + j];
        float go = grow[96 + j]   + (float)xgrow[3 * HID + j];
        float it = sigmoidf_(gi);
        float ft = sigmoidf_(gf);
        float gt = tanhf(gg);
        float ot = sigmoidf_(go);
        float c  = ft * crow[j] + it * gt;
        crow[j]  = c;
        hrow[j]  = (__bf16)(ot * tanhf(c));
      }
    }

    // ---- device-scope barrier (monotonic counter; reset by k_init)
    __syncthreads();
    if (tid == 0) {
      __threadfence();
      __hip_atomic_fetch_add(bar, 1, __ATOMIC_RELAXED, __HIP_MEMORY_SCOPE_AGENT);
      int target = NWG * (t + 1);
      while (__hip_atomic_load(bar, __ATOMIC_RELAXED, __HIP_MEMORY_SCOPE_AGENT) <
             target) {
        __builtin_amdgcn_s_sleep(1);
      }
      __threadfence();
    }
    __syncthreads();
  }
}

// ---------------- head GEMM: out = h_last @ W^T + b ----------------
__global__ __launch_bounds__(256)
void k_head(const __bf16* __restrict__ hlast, const __bf16* __restrict__ Wb,
            const float* __restrict__ bias, float* __restrict__ out) {
  const int lane = threadIdx.x & 31, wv = threadIdx.x >> 5;
  const int ln = lane & 15, lh = lane >> 4;
  const int n = blockIdx.x * 128 + wv * 16 + ln;

  v8f acc[4];
#pragma unroll
  for (int mt = 0; mt < 4; ++mt) acc[mt] = fzero8();

  for (int kc = 0; kc < HID; kc += 32) {
    v16bf bb = *(const v16bf*)(Wb + (size_t)n * HID + kc + (lh << 4));
#pragma unroll
    for (int mt = 0; mt < 4; ++mt) {
      v16bf a = load_afrag_g(hlast, mt * 16, HID, kc, ln, lh);
      acc[mt] = wmma_bf16(a, bb, acc[mt]);
    }
  }
  if (n < NCLS) {
    float bv = bias[n];
#pragma unroll
    for (int mt = 0; mt < 4; ++mt)
#pragma unroll
      for (int r = 0; r < 8; ++r) {
        int m = mt * 16 + lh * 8 + r;
        out[(size_t)m * NCLS + n] = acc[mt][r] + bv;
      }
  }
}

// ---------------- host launcher ----------------
extern "C" void kernel_launch(void* const* d_in, const int* in_sizes, int n_in,
                              void* d_out, int out_size, void* d_ws, size_t ws_size,
                              hipStream_t stream) {
  (void)in_sizes; (void)n_in; (void)out_size; (void)ws_size;
  const int*   X     = (const int*)d_in[0];
  const float* emb   = (const float*)d_in[1];
  const float* w_ih0 = (const float*)d_in[2];
  const float* w_hh0 = (const float*)d_in[3];
  const float* b_ih0 = (const float*)d_in[4];
  const float* b_hh0 = (const float*)d_in[5];
  const float* w_ih1 = (const float*)d_in[6];
  const float* w_hh1 = (const float*)d_in[7];
  const float* b_ih1 = (const float*)d_in[8];
  const float* b_hh1 = (const float*)d_in[9];
  const float* Wf    = (const float*)d_in[10];
  const float* bf    = (const float*)d_in[11];
  float* out = (float*)d_out;

  char* ws = (char*)d_ws;
  size_t off = 0;
  auto take = [&](size_t bytes) -> char* {
    char* p = ws + off;
    off += (bytes + 255) & ~(size_t)255;
    return p;
  };
  __bf16* wih0b = (__bf16*)take((size_t)GATES * HID * 2);
  __bf16* whh0b = (__bf16*)take((size_t)GATES * HID * 2);
  __bf16* wih1b = (__bf16*)take((size_t)GATES * HID * 2);
  __bf16* whh1b = (__bf16*)take((size_t)GATES * HID * 2);
  __bf16* Wb    = (__bf16*)take((size_t)NCLSP * HID * 2);
  float*  bias0 = (float*)take(GATES * 4);
  float*  bias1 = (float*)take(GATES * 4);
  __bf16* xemb  = (__bf16*)take((size_t)BSZ * TLEN * HID * 2);
  __bf16* xg    = (__bf16*)take((size_t)BSZ * TLEN * GATES * 2);
  __bf16* h0seq = (__bf16*)take((size_t)(TLEN + 1) * BSZ * HID * 2);
  __bf16* h1seq = (__bf16*)take((size_t)(TLEN + 1) * BSZ * HID * 2);
  int*    bar   = (int*)take(256);

  (void)hipFuncSetAttribute(reinterpret_cast<const void*>(k_scan),
                            hipFuncAttributeMaxDynamicSharedMemorySize, SMEM_B);

  k_init<<<1, 256, 0, stream>>>(h0seq, h1seq, bar);

  const int WN = GATES * HID;                 // 1,048,576
  k_cast<<<WN / (256 * 8), 256, 0, stream>>>(w_ih0, wih0b, WN, WN);
  k_cast<<<WN / (256 * 8), 256, 0, stream>>>(w_hh0, whh0b, WN, WN);
  k_cast<<<WN / (256 * 8), 256, 0, stream>>>(w_ih1, wih1b, WN, WN);
  k_cast<<<WN / (256 * 8), 256, 0, stream>>>(w_hh1, whh1b, WN, WN);
  const int WPN = NCLSP * HID;                // padded
  k_cast<<<WPN / (256 * 8), 256, 0, stream>>>(Wf, Wb, NCLS * HID, WPN);

  k_bias<<<1, 256, 0, stream>>>(b_ih0, b_hh0, bias0);
  k_bias<<<1, 256, 0, stream>>>(b_ih1, b_hh1, bias1);

  k_embed<<<(BSZ * TLEN * HID / 8) / 256, 256, 0, stream>>>(X, emb, xemb);

  // layer 0
  k_gemm16k<<<dim3(GATES / 256, BSZ * TLEN / 64), 256, 0, stream>>>(
      xemb, wih0b, bias0, xg);
  k_scan<<<NWG, 256, SMEM_B, stream>>>(xg, h0seq, whh0b, bar);

  // layer 1 (input = h0 blocks 1..T)
  k_gemm16k<<<dim3(GATES / 256, BSZ * TLEN / 64), 256, 0, stream>>>(
      h0seq + (size_t)BSZ * HID, wih1b, bias1, xg);
  k_scan<<<NWG, 256, SMEM_B, stream>>>(xg, h1seq, whh1b, bar + 1);

  // head
  k_head<<<NCLSP / 128, 256, 0, stream>>>(
      h1seq + (size_t)TLEN * BSZ * HID, Wb, bf, out);
}
